// MaxViTBlock_31963146616892
// MI455X (gfx1250) — matrix operand order repaired and checked by reading, not compiled, not run on hardware
//
#include <hip/hip_runtime.h>
#include <cmath>

// ---------------------------------------------------------------------------
// MaxViT block for MI455X (gfx1250, wave32, WMMA).
// Master activation: fp32 NHWC [M=100352, C]. All big GEMMs: bf16 WMMA
// (v_wmma_f32_16x16x32_bf16) with fp32 accumulate, fused epilogues.
// GEMM v2: 128x128x32 tiles, 8 waves, double-buffered LDS fed by
// GLOBAL_LOAD_ASYNC_TO_LDS_B128 (ASYNCcnt + s_wait_asynccnt).
// Workspace requirement: ~569.4 MB (see offsets in kernel_launch).
// ---------------------------------------------------------------------------

typedef __bf16 bf16;
typedef __bf16 v16bf __attribute__((ext_vector_type(16)));
typedef __bf16 v8bf  __attribute__((ext_vector_type(8)));
typedef float  v8f   __attribute__((ext_vector_type(8)));

#define DEV __device__ __forceinline__

DEV float gelu_exact(float x) { return 0.5f * x * (1.0f + erff(x * 0.70710678118654752f)); }

// Async global->LDS copy of 16 bytes (CDNA5 Tensor/async path, ASYNCcnt).
// lds_byte: LDS byte address (= low 32 bits of generic shared address).
DEV void async_b128(unsigned lds_byte, const void* gaddr) {
  asm volatile("global_load_async_to_lds_b128 %0, %1, off"
               :: "v"(lds_byte), "v"(gaddr) : "memory");
}
DEV void wait_async0() { asm volatile("s_wait_asynccnt 0x0" ::: "memory"); }

// ---------------- NCHW -> NHWC (+ bn0-folded bf16 copy) --------------------
__global__ void k_prep(const float* __restrict__ x, const float* __restrict__ s0,
                       const float* __restrict__ b0, float* __restrict__ A,
                       bf16* __restrict__ Bh) {
  __shared__ float t[16][17];
  int b = blockIdx.z, hw0 = blockIdx.x * 16, c0 = blockIdx.y * 16;
  int tx = threadIdx.x, ty = threadIdx.y;
  t[ty][tx] = x[((size_t)(b * 256 + c0 + ty)) * 3136 + hw0 + tx];
  __syncthreads();
  int c = c0 + tx;
  float v = t[tx][ty];
  size_t o = ((size_t)(b * 3136 + hw0 + ty)) * 256 + c;
  A[o] = v;
  Bh[o] = (bf16)(v * s0[c] + b0[c]);
}

// ---------------- NHWC -> NCHW (final output) ------------------------------
__global__ void k_out(const float* __restrict__ A, float* __restrict__ out) {
  __shared__ float t[16][17];
  int b = blockIdx.z, hw0 = blockIdx.x * 16, c0 = blockIdx.y * 16;
  int tx = threadIdx.x, ty = threadIdx.y;
  t[ty][tx] = A[((size_t)(b * 3136 + hw0 + ty)) * 256 + c0 + tx];
  __syncthreads();
  out[((size_t)(b * 256 + c0 + ty)) * 3136 + hw0 + tx] = t[tx][ty];
}

__global__ void k_zero(float* p, int n) {
  int i = blockIdx.x * 256 + threadIdx.x;
  if (i < n) p[i] = 0.0f;
}

__global__ void k_cvtw(const float* __restrict__ s, bf16* __restrict__ d, int n) {
  int i = blockIdx.x * 256 + threadIdx.x;
  if (i < n) d[i] = (bf16)s[i];
}

// ---------------- WMMA GEMM: out[M,N] = act(A[M,K] * W[N,K]^T ...) ----------
// A: bf16 row-major. W: bf16 [N][K] (pre-converted). BM=BN=128, BK=32,
// 256 threads = 8 waves; wave (wr=wv&3, wc=wv>>2) owns a 32x64 sub-tile:
// 2 A-frags x 4 B-frags -> 8 WMMA per k-step.
// ASYNC: double-buffered LDS fed by global_load_async_to_lds_b128.
// ASCALE (SE fold): synchronous staging, A[m,k] *= ascale[(m/3136)*K + k].
// Epilogue: scl/sh = per-N scale+shift (BN fold) or bias-only (scl=null);
// ACT=1 -> exact GELU; optional fp32 residual; fp32 and/or bf16 outputs.
template <int ACT, bool ASCALE, bool ASYNC>
__global__ __launch_bounds__(256) void k_gemm(
    const bf16* __restrict__ Abuf, const float* __restrict__ ascale,
    const bf16* __restrict__ W, const float* __restrict__ scl,
    const float* __restrict__ sh, const float* __restrict__ resid,
    float* __restrict__ outF, bf16* __restrict__ outH, int M, int N, int K) {
  __shared__ bf16 As[2][128 * 32];
  __shared__ bf16 Bs[2][128 * 32];
  int m0 = blockIdx.x * 128, n0 = blockIdx.y * 128;
  int tid = threadIdx.x, lane = tid & 31, wv = tid >> 5;
  int wr = wv & 3, wc = wv >> 2;
  int lr = tid >> 1, lk = (tid & 1) * 16;  // staging: row 0..127, k-seg 0/16
  v8f acc[2][4] = {};
  v16bf af[2];

  auto issue = [&](int kk, int buf) {  // async global->LDS: 2xB128 for A & B
    const bf16* ga = Abuf + (size_t)(m0 + lr) * K + kk + lk;
    unsigned la = (unsigned)(uintptr_t)&As[buf][lr * 32 + lk];
    async_b128(la, ga);
    async_b128(la + 16, ga + 8);
    const bf16* gb = W + (size_t)(n0 + lr) * K + kk + lk;
    unsigned lb = (unsigned)(uintptr_t)&Bs[buf][lr * 32 + lk];
    async_b128(lb, gb);
    async_b128(lb + 16, gb + 8);
  };

  auto compute = [&](int buf) {
#pragma unroll
    for (int mf = 0; mf < 2; ++mf) {
      // A fragment 16x32: lane = row m, contiguous 16-elem K run
      const bf16* arow =
          &As[buf][(wr * 32 + mf * 16 + (lane & 15)) * 32 + ((lane >> 4) << 4)];
      v8bf alo = *(const v8bf*)arow;
      v8bf ahi = *(const v8bf*)(arow + 8);
#pragma unroll
      for (int e = 0; e < 8; ++e) { af[mf][e] = alo[e]; af[mf][e + 8] = ahi[e]; }
    }
#pragma unroll
    for (int t = 0; t < 4; ++t) {
      // B fragment 32x16: lane = column n, contiguous 16-elem K run
      const bf16* brow =
          &Bs[buf][(wc * 64 + t * 16 + (lane & 15)) * 32 + ((lane >> 4) << 4)];
      v16bf bfr = *(const v16bf*)brow;
      acc[0][t] = __builtin_amdgcn_wmma_f32_16x16x32_bf16(
          false, af[0], false, bfr, (short)0, acc[0][t], false, false);
      acc[1][t] = __builtin_amdgcn_wmma_f32_16x16x32_bf16(
          false, af[1], false, bfr, (short)0, acc[1][t], false, false);
    }
  };

  if (ASYNC) {
    issue(0, 0);
    int buf = 0;
    for (int kk = 0; kk < K; kk += 32) {
      wait_async0();     // my async loads for `buf` landed in LDS
      __syncthreads();   // everyone's landed; prior reads of buf^1 done
      if (kk + 32 < K) issue(kk + 32, buf ^ 1);
      compute(buf);
      buf ^= 1;
    }
  } else {
    for (int kk = 0; kk < K; kk += 32) {
      {  // A tile with SE per-(batch,channel) scale folded in
        const bf16* g = Abuf + (size_t)(m0 + lr) * K + kk + lk;
        if (ASCALE) {
          int bat = (m0 + lr) / 3136;
          const float* se = ascale + (size_t)bat * K + kk + lk;
#pragma unroll
          for (int i = 0; i < 16; ++i)
            As[0][lr * 32 + lk + i] = (bf16)((float)g[i] * se[i]);
        } else {
          *(v16bf*)&As[0][lr * 32 + lk] = *(const v16bf*)g;
        }
      }
      *(v16bf*)&Bs[0][lr * 32 + lk] =
          *(const v16bf*)(W + (size_t)(n0 + lr) * K + kk + lk);
      __syncthreads();
      compute(0);
      __syncthreads();
    }
  }

  // ---- fused epilogue; C/D layout: elem v -> m = v + 8*(lane/16), n = lane%16
#pragma unroll
  for (int mf = 0; mf < 2; ++mf) {
#pragma unroll
    for (int t = 0; t < 4; ++t) {
#pragma unroll
      for (int v = 0; v < 8; ++v) {
        int m = m0 + wr * 32 + mf * 16 + v + ((lane >> 4) << 3);
        int n = n0 + wc * 64 + t * 16 + (lane & 15);
        float o = acc[mf][t][v];
        if (scl)      o = o * scl[n] + sh[n];
        else if (sh)  o += sh[n];
        if (ACT == 1) o = gelu_exact(o);
        size_t idx = (size_t)m * N + n;
        if (resid) o += resid[idx];
        if (outF) outF[idx] = o;
        if (outH) outH[idx] = (bf16)o;
      }
    }
  }
}

// ---------------- depthwise 3x3 (NHWC) + bn+GELU + SE partial sums ----------
__global__ __launch_bounds__(256) void k_dw(const bf16* __restrict__ h1,
                                            const float* __restrict__ w9,
                                            const float* __restrict__ s,
                                            const float* __restrict__ bsh,
                                            bf16* __restrict__ h2,
                                            float* __restrict__ se_sum) {
  int blk = blockIdx.x;
  int b = blk / 392, chunk = blk % 392;  // 8 tokens per block
  int c0 = threadIdx.x * 4;              // 4 channels per thread
  float wt[4][9], bns[4], bnb[4], ss[4] = {0, 0, 0, 0};
#pragma unroll
  for (int j = 0; j < 4; ++j) {
#pragma unroll
    for (int q = 0; q < 9; ++q) wt[j][q] = w9[(c0 + j) * 9 + q];
    bns[j] = s[c0 + j]; bnb[j] = bsh[c0 + j];
  }
  for (int t = 0; t < 8; ++t) {
    int hw = chunk * 8 + t, hh = hw / 56, ww = hw % 56;
    float acc[4] = {0, 0, 0, 0};
    for (int dh = -1; dh <= 1; ++dh) {
      int y = hh + dh; if (y < 0 || y >= 56) continue;
      for (int dw = -1; dw <= 1; ++dw) {
        int xw = ww + dw; if (xw < 0 || xw >= 56) continue;
        const bf16* p = h1 + ((size_t)(b * 3136 + y * 56 + xw)) * 1024 + c0;
        int q = (dh + 1) * 3 + (dw + 1);
#pragma unroll
        for (int j = 0; j < 4; ++j) acc[j] += (float)p[j] * wt[j][q];
      }
    }
    bf16* po = h2 + ((size_t)(b * 3136 + hw)) * 1024 + c0;
#pragma unroll
    for (int j = 0; j < 4; ++j) {
      float o = gelu_exact(acc[j] * bns[j] + bnb[j]);
      po[j] = (bf16)o;
      ss[j] += o;
    }
  }
#pragma unroll
  for (int j = 0; j < 4; ++j) atomicAdd(&se_sum[b * 1024 + c0 + j], ss[j]);
}

// ---------------- squeeze-excite MLP (one block per batch) ------------------
__global__ __launch_bounds__(256) void k_se(const float* __restrict__ sums,
                                            const float* __restrict__ w1,
                                            const float* __restrict__ b1,
                                            const float* __restrict__ w2,
                                            const float* __restrict__ b2,
                                            float* __restrict__ scale) {
  __shared__ float mean[1024];
  __shared__ float hid[256];
  int b = blockIdx.x, tid = threadIdx.x;
  for (int c = tid; c < 1024; c += 256) mean[c] = sums[b * 1024 + c] * (1.0f / 3136.0f);
  __syncthreads();
  {
    float a = 0.0f;
    const float* wr = w1 + (size_t)tid * 1024;
    for (int c = 0; c < 1024; ++c) a += mean[c] * wr[c];
    a += b1[tid];
    hid[tid] = a > 0.0f ? a : 0.0f;
  }
  __syncthreads();
  for (int c = tid; c < 1024; c += 256) {
    float a = 0.0f;
    const float* wr = w2 + (size_t)c * 256;
    for (int j = 0; j < 256; ++j) a += hid[j] * wr[j];
    a += b2[c];
    scale[b * 1024 + c] = 1.0f / (1.0f + expf(-a));
  }
}

// ---------------- LayerNorm over 256 channels, 1 wave / token ---------------
template <typename Tin>
__global__ __launch_bounds__(256) void k_ln(const Tin* __restrict__ in,
                                            const float* __restrict__ g,
                                            const float* __restrict__ bta,
                                            float* __restrict__ outF,
                                            bf16* __restrict__ outH) {
  int wv = threadIdx.x >> 5, lane = threadIdx.x & 31;
  size_t m = (size_t)blockIdx.x * 8 + wv;
  const Tin* p = in + m * 256;
  float v[8], s = 0.0f;
#pragma unroll
  for (int j = 0; j < 8; ++j) { v[j] = (float)p[j * 32 + lane]; s += v[j]; }
#pragma unroll
  for (int o = 16; o; o >>= 1) s += __shfl_xor(s, o, 32);
  float mu = s * (1.0f / 256.0f), q = 0.0f;
#pragma unroll
  for (int j = 0; j < 8; ++j) { float d = v[j] - mu; q += d * d; }
#pragma unroll
  for (int o = 16; o; o >>= 1) q += __shfl_xor(q, o, 32);
  float inv = rsqrtf(q * (1.0f / 256.0f) + 1e-5f);
#pragma unroll
  for (int j = 0; j < 8; ++j) {
    int c = j * 32 + lane;
    float o = (v[j] - mu) * inv * g[c] + bta[c];
    if (outF) outF[m * 256 + c] = o;
    if (outH) outH[m * 256 + c] = (bf16)o;
  }
}

// ---------------- window/grid attention core: 1 wave per (window, head) -----
// qkv bf16 [M][768] token-major; out bf16 [M][256]; rel-pos bias table [169][8].
__global__ __launch_bounds__(32) void k_attn(const bf16* __restrict__ qkv,
                                             const float* __restrict__ btab,
                                             bf16* __restrict__ out, int gridmode) {
  __shared__ float q[49][33], kk[49][33], vv[49][33];  // pad 33: conflict-free
  __shared__ float sc[49 * 49];
  int head = blockIdx.x & 7, win = blockIdx.x >> 3;
  int lane = threadIdx.x;
  int b = win >> 6, wr = (win >> 3) & 7, wc = win & 7;
  const float qs = 0.17677669529663687f;  // 1/sqrt(32)

  auto tok = [&](int p) {
    int ih = p / 7, iw = p % 7;
    int h = gridmode ? (ih * 8 + wr) : (wr * 7 + ih);
    int w = gridmode ? (iw * 8 + wc) : (wc * 7 + iw);
    return (size_t)(b * 3136 + h * 56 + w);
  };

  for (int p = 0; p < 49; ++p) {
    const bf16* base = qkv + tok(p) * 768 + head * 32 + lane;
    q[p][lane] = (float)base[0] * qs;
    kk[p][lane] = (float)base[256];
    vv[p][lane] = (float)base[512];
  }
  __syncthreads();

  for (int i = 0; i < 49; ++i) {
    int ih = i / 7, iw = i % 7;
    for (int j = lane; j < 49; j += 32) {
      float a = 0.0f;
#pragma unroll
      for (int d = 0; d < 32; ++d) a += q[i][d] * kk[j][d];
      int jh = j / 7, jw = j % 7;
      a += btab[((ih - jh + 6) * 13 + (iw - jw + 6)) * 8 + head];
      sc[i * 49 + j] = a;
    }
  }
  __syncthreads();

  for (int i = lane; i < 49; i += 32) {  // row softmax, lane-per-row
    float mx = -1e30f;
    for (int j = 0; j < 49; ++j) mx = fmaxf(mx, sc[i * 49 + j]);
    float sm = 0.0f;
    for (int j = 0; j < 49; ++j) { float e = __expf(sc[i * 49 + j] - mx); sc[i * 49 + j] = e; sm += e; }
    float r = 1.0f / sm;
    for (int j = 0; j < 49; ++j) sc[i * 49 + j] *= r;
  }
  __syncthreads();

  for (int i = 0; i < 49; ++i) {
    float o = 0.0f;
    for (int j = 0; j < 49; ++j) o += sc[i * 49 + j] * vv[j][lane];
    out[tok(i) * 256 + head * 32 + lane] = (bf16)o;
  }
}

// ---------------------------------------------------------------------------
extern "C" void kernel_launch(void* const* d_in, const int* in_sizes, int n_in,
                              void* d_out, int out_size, void* d_ws, size_t ws_size,
                              hipStream_t stream) {
  const float* x = (const float*)d_in[0];
  const float* bn0_s = (const float*)d_in[1];
  const float* bn0_b = (const float*)d_in[2];
  const float* pw_w = (const float*)d_in[3];
  const float* pw_s = (const float*)d_in[4];
  const float* pw_b = (const float*)d_in[5];
  const float* dw_w = (const float*)d_in[6];
  const float* dw_s = (const float*)d_in[7];
  const float* dw_b = (const float*)d_in[8];
  const float* se_w1 = (const float*)d_in[9];
  const float* se_b1 = (const float*)d_in[10];
  const float* se_w2 = (const float*)d_in[11];
  const float* se_b2 = (const float*)d_in[12];
  const float* pr_w = (const float*)d_in[13];
  const float* pr_s = (const float*)d_in[14];
  const float* pr_b = (const float*)d_in[15];
  const float* p_ln1_g[2] = {(const float*)d_in[16], (const float*)d_in[27]};
  const float* p_ln1_b[2] = {(const float*)d_in[17], (const float*)d_in[28]};
  const float* p_qkv_w[2] = {(const float*)d_in[18], (const float*)d_in[29]};
  const float* p_qkv_b[2] = {(const float*)d_in[19], (const float*)d_in[30]};
  const float* p_bias[2]  = {(const float*)d_in[20], (const float*)d_in[31]};
  const float* p_ln2_g[2] = {(const float*)d_in[21], (const float*)d_in[32]};
  const float* p_ln2_b[2] = {(const float*)d_in[22], (const float*)d_in[33]};
  const float* p_m_w1[2]  = {(const float*)d_in[23], (const float*)d_in[34]};
  const float* p_m_b1[2]  = {(const float*)d_in[24], (const float*)d_in[35]};
  const float* p_m_w2[2]  = {(const float*)d_in[25], (const float*)d_in[36]};
  const float* p_m_b2[2]  = {(const float*)d_in[26], (const float*)d_in[37]};

  const int M = 100352;  // 32 * 56 * 56 tokens
  char* ws = (char*)d_ws;
  // Workspace layout (all offsets 256B aligned); total 569,376,768 bytes.
  float* A      = (float*)ws;                        // fp32 NHWC main act [M,256]  102.76 MB
  bf16*  Bh     = (bf16*)(ws + 102760448);           // bf16 GEMM-in       [M,256]   51.38 MB
  bf16*  Cb     = (bf16*)(ws + 154140672);           // bf16 wide scratch  [M,1024] 205.52 MB
  bf16*  Db     = (bf16*)(ws + 359661568);           // bf16 wide scratch2 (dw out / qkv)
  bf16*  AttnO  = (bf16*)(ws + 513802240);           // bf16 attn out      [M,256]  (tail of Db)
  float* se_sum = (float*)(ws + 565182464);          // [32,1024]
  float* se_scl = se_sum + 32 * 1024;                // [32,1024]
  bf16*  wb     = (bf16*)(ws + 565444608);           // bf16 weight pool (3.93 MB)
  bf16* wpw = wb;                 // [1024][256]
  bf16* wpr = wb + 262144;        // [256][1024]
  bf16* wqkv[2] = {wb + 524288, wb + 720896};        // [768][256]
  bf16* wm1[2]  = {wb + 917504, wb + 1179648};       // [1024][256]
  bf16* wm2[2]  = {wb + 1441792, wb + 1703936};      // [256][1024]

  auto cvt = [&](const float* s, bf16* d, int n) {
    k_cvtw<<<(n + 255) / 256, 256, 0, stream>>>(s, d, n);
  };
  cvt(pw_w, wpw, 262144);
  cvt(pr_w, wpr, 262144);
  for (int ph = 0; ph < 2; ++ph) {
    cvt(p_qkv_w[ph], wqkv[ph], 196608);
    cvt(p_m_w1[ph], wm1[ph], 262144);
    cvt(p_m_w2[ph], wm2[ph], 262144);
  }

  dim3 tb16(16, 16);

  // ---------------- MBConv ----------------
  k_prep<<<dim3(196, 16, 32), tb16, 0, stream>>>(x, bn0_s, bn0_b, A, Bh);
  k_gemm<1, false, true><<<dim3(M / 128, 1024 / 128), 256, 0, stream>>>(
      Bh, nullptr, wpw, pw_s, pw_b, nullptr, nullptr, Cb, M, 1024, 256);
  k_zero<<<128, 256, 0, stream>>>(se_sum, 32 * 1024);
  k_dw<<<32 * 392, 256, 0, stream>>>(Cb, dw_w, dw_s, dw_b, Db, se_sum);
  k_se<<<32, 256, 0, stream>>>(se_sum, se_w1, se_b1, se_w2, se_b2, se_scl);
  k_gemm<1, true, false><<<dim3(M / 128, 256 / 128), 256, 0, stream>>>(
      Db, se_scl, wpr, pr_s, pr_b, A, A, nullptr, M, 256, 1024);  // +residual -> A

  // ---------------- block (window) attention, then grid attention ----------
  for (int ph = 0; ph < 2; ++ph) {
    k_ln<float><<<M / 8, 256, 0, stream>>>(A, p_ln1_g[ph], p_ln1_b[ph], nullptr, Bh);
    k_gemm<0, false, true><<<dim3(M / 128, 768 / 128), 256, 0, stream>>>(
        Bh, nullptr, wqkv[ph], nullptr, p_qkv_b[ph], nullptr, nullptr, Db, M, 768, 256);
    k_attn<<<2048 * 8, 32, 0, stream>>>(Db, p_bias[ph], AttnO, ph);
    // x = rwp(ln2(attn)) replaces main activation (faithful: no proj, no resid)
    k_ln<bf16><<<M / 8, 256, 0, stream>>>(AttnO, p_ln2_g[ph], p_ln2_b[ph], A, Bh);
    // MLP with residual: x = x + W2(gelu(W1 x + b1)) + b2
    k_gemm<1, false, true><<<dim3(M / 128, 1024 / 128), 256, 0, stream>>>(
        Bh, nullptr, wm1[ph], nullptr, p_m_b1[ph], nullptr, nullptr, Cb, M, 1024, 256);
    k_gemm<0, false, true><<<dim3(M / 128, 256 / 128), 256, 0, stream>>>(
        Cb, nullptr, wm2[ph], nullptr, p_m_b2[ph], A, A, nullptr, M, 256, 1024);
  }

  k_out<<<dim3(196, 16, 32), tb16, 0, stream>>>(A, (float*)d_out);
}
